// Model_13391708029666
// MI455X (gfx1250) — matrix operand order repaired and checked by reading, not compiled, not run on hardware
//
#include <hip/hip_runtime.h>
#include <hip/hip_bf16.h>
#include <math.h>
#include <cstdint>

// ---------------------------------------------------------------------------
// Galerkin transformer forward for MI455X (gfx1250), wave32 + WMMA bf16.
// GEMMs: v_wmma_f32_16x16x32_bf16 with double-buffered LDS tiles filled by
// global_load_async_to_lds_b128 (ASYNCcnt-tracked DMA, no VGPR round-trip).
// ---------------------------------------------------------------------------

typedef __attribute__((ext_vector_type(16))) __bf16 v16bf;
typedef __attribute__((ext_vector_type(8)))  float  v8f;

#define NT     65536   // total tokens  (B * H * W)
#define NPB    16384   // tokens per batch (128*128)
#define DMODEL 256
#define NH     8
#define DH     32

struct U32B { uint4 a, b; };   // 32 bytes == one v16bf fragment

__device__ __forceinline__ float bf2f(unsigned short h) {
  unsigned u = ((unsigned)h) << 16;
  return __builtin_bit_cast(float, u);
}
__device__ __forceinline__ unsigned short f2bf(float f) {
  unsigned u = __builtin_bit_cast(unsigned, f);
  unsigned r = u + 0x7FFFu + ((u >> 16) & 1u);   // round-to-nearest-even
  return (unsigned short)(r >> 16);
}
__device__ __forceinline__ float gelu_exact(float x) {
  return 0.5f * x * (1.0f + erff(x * 0.70710678118654752f));
}

// Async global -> LDS 16B copy (GV mode). INST_OFFSET is added to BOTH the
// LDS and global addresses per the ISA, so one base covers a 32B pair.
__device__ __forceinline__ void async_copy16(unsigned lds_addr,
                                             unsigned long long gaddr) {
  asm volatile("global_load_async_to_lds_b128 %0, %1, off"
               :: "v"(lds_addr), "v"(gaddr) : "memory");
}
__device__ __forceinline__ void async_copy32(unsigned lds_addr,
                                             unsigned long long gaddr) {
  asm volatile("global_load_async_to_lds_b128 %0, %1, off\n\t"
               "global_load_async_to_lds_b128 %0, %1, off offset:16"
               :: "v"(lds_addr), "v"(gaddr) : "memory");
}
__device__ __forceinline__ unsigned lds_addr_of(const void* p) {
  return (unsigned)(uintptr_t)p;   // LDS aperture: addr[31:0] is the LDS offset
}

// ---------------------------------------------------------------------------
// Utility kernels
// ---------------------------------------------------------------------------
__global__ void k_f32_to_bf16(const float* __restrict__ in,
                              unsigned short* __restrict__ out, int n) {
  int i = blockIdx.x * blockDim.x + threadIdx.x;
  if (i < n) out[i] = f2bf(in[i]);
}

// Convert [L][K][M] f32 weights to [L][M][K] bf16 (K-major for GEMM B tiles).
__global__ void k_f32_to_bf16_T(const float* __restrict__ in,
                                unsigned short* __restrict__ out,
                                int K, int M, int total) {
  int i = blockIdx.x * blockDim.x + threadIdx.x;
  if (i >= total) return;
  int le = K * M;
  int l = i / le;
  int r = i % le;
  int m = r / K;
  int k = r % K;
  out[i] = f2bf(in[(size_t)l * le + (size_t)k * M + m]);
}

__global__ void k_zero_f32(float* __restrict__ p, int n) {
  int i = blockIdx.x * blockDim.x + threadIdx.x;
  if (i < n) p[i] = 0.0f;
}

// ---------------------------------------------------------------------------
// Embed: t[n,d] = concat(x, coords)[n,:] @ embed_w + embed_b   (in_dim = 5)
// ---------------------------------------------------------------------------
__global__ __launch_bounds__(256) void k_embed(
    const float* __restrict__ x, const float* __restrict__ coords,
    const float* __restrict__ ew, const float* __restrict__ eb,
    float* __restrict__ t) {
  int idx = blockIdx.x * 256 + threadIdx.x;   // NT * 256 threads
  int d = idx & 255;
  int n = idx >> 8;
  int b = n / NPB;
  int hw = n % NPB;
  float i0 = x[(size_t)(b * 3 + 0) * NPB + hw];
  float i1 = x[(size_t)(b * 3 + 1) * NPB + hw];
  float i2 = x[(size_t)(b * 3 + 2) * NPB + hw];
  float i3 = coords[hw];
  float i4 = coords[NPB + hw];
  float acc = eb[d];
  acc += i0 * ew[0 * DMODEL + d];
  acc += i1 * ew[1 * DMODEL + d];
  acc += i2 * ew[2 * DMODEL + d];
  acc += i3 * ew[3 * DMODEL + d];
  acc += i4 * ew[4 * DMODEL + d];
  t[(size_t)n * DMODEL + d] = acc;
}

// ---------------------------------------------------------------------------
// Row LayerNorm over D=256; one wave (32 lanes) per row, 8 rows per block.
// ---------------------------------------------------------------------------
template <bool OUT_BF16>
__global__ __launch_bounds__(256) void k_layernorm(
    const float* __restrict__ in, const float* __restrict__ g,
    const float* __restrict__ bta, void* __restrict__ out) {
  const int lane = threadIdx.x & 31;
  const int row  = blockIdx.x * 8 + (threadIdx.x >> 5);
  const float* p = in + (size_t)row * DMODEL + lane * 8;
  float v[8];
#pragma unroll
  for (int k = 0; k < 8; ++k) v[k] = p[k];
  float s = 0.0f;
#pragma unroll
  for (int k = 0; k < 8; ++k) s += v[k];
#pragma unroll
  for (int m = 16; m > 0; m >>= 1) s += __shfl_xor(s, m, 32);
  float mean = s * (1.0f / 256.0f);
  float q = 0.0f;
#pragma unroll
  for (int k = 0; k < 8; ++k) { float d = v[k] - mean; q += d * d; }
#pragma unroll
  for (int m = 16; m > 0; m >>= 1) q += __shfl_xor(q, m, 32);
  float rs = rsqrtf(q * (1.0f / 256.0f) + 1e-5f);
#pragma unroll
  for (int k = 0; k < 8; ++k) {
    int d = lane * 8 + k;
    float o = (v[k] - mean) * rs * g[d] + bta[d];
    if constexpr (OUT_BF16)
      ((unsigned short*)out)[(size_t)row * DMODEL + d] = f2bf(o);
    else
      ((float*)out)[(size_t)row * DMODEL + d] = o;
  }
}

// ---------------------------------------------------------------------------
// WMMA bf16 GEMM:  out[N,M] = A[N,K] @ W[K,M]  (+bias) (gelu) (+res f32)
// W is supplied pre-transposed as Wt[M][K].
// Block tile 128x64, 8 waves, wave tile 32x32 = 2x2 wmma 16x16x32 tiles.
// Double-buffered LDS tiles, filled by async global->LDS DMA: the next
// k-tile streams in (ASYNCcnt) while the current one feeds the WMMAs.
// ---------------------------------------------------------------------------
template <bool BIAS, bool GELU_, bool RES, bool OUT_BF16>
__global__ __launch_bounds__(256) void k_gemm_bf16(
    const unsigned short* __restrict__ A,    // [N,K]  bf16
    const unsigned short* __restrict__ Wt,   // [M,K]  bf16 (transposed)
    const float* __restrict__ bias,          // [M] or null
    const float* __restrict__ res,           // [N,M] or null
    void* __restrict__ out, int K, int M) {
  __shared__ unsigned short As[2][128][40];  // 128 x 32 tile (+8 pad)  x2
  __shared__ unsigned short Bs[2][64][40];   // 64 cols x 32 k (+8 pad) x2

  const int tid  = threadIdx.x;
  const int lane = tid & 31;
  const int wv   = tid >> 5;
  const int wi   = wv & 3;     // 4 row-waves  -> 32 rows each
  const int wj   = wv >> 2;    // 2 col-waves  -> 32 cols each
  const int row0 = blockIdx.y * 128;
  const int col0 = blockIdx.x * 64;

  // staging coordinates for this thread (3 async b128 ops per stage)
  const int ar = (tid * 16) >> 5;        // A row      (0..127)
  const int ac = (tid * 16) & 31;        // A k-offset (0 or 16), 32 elems via pair
  const int bj = (tid * 8) >> 5;         // B col      (0..63)
  const int bk = (tid * 8) & 31;         // B k-offset (0,8,16,24)

  auto stage = [&](int buf, int k0) {
    async_copy32(lds_addr_of(&As[buf][ar][ac]),
                 (unsigned long long)(uintptr_t)(A + (size_t)(row0 + ar) * K + k0 + ac));
    async_copy16(lds_addr_of(&Bs[buf][bj][bk]),
                 (unsigned long long)(uintptr_t)(Wt + (size_t)(col0 + bj) * K + k0 + bk));
  };

  v8f acc[2][2];
#pragma unroll
  for (int h = 0; h < 2; ++h)
#pragma unroll
    for (int g2 = 0; g2 < 2; ++g2)
#pragma unroll
      for (int j = 0; j < 8; ++j) acc[h][g2][j] = 0.0f;

  stage(0, 0);
  int buf = 0;
  for (int k0 = 0; k0 < K; k0 += 32) {
    const bool has_next = (k0 + 32) < K;
    if (has_next) {
      stage(buf ^ 1, k0 + 32);
      // 3 newest (next tile) may stay in flight; everything older is done.
      asm volatile("s_wait_asynccnt 0x3" ::: "memory");
    } else {
      asm volatile("s_wait_asynccnt 0x0" ::: "memory");
    }
    __syncthreads();

#pragma unroll
    for (int h = 0; h < 2; ++h) {
      // A fragment (16x32): lanes 0-15 K{0..7,16..23}, lanes 16-31 K{8..15,24..31}
      int rr = 32 * wi + 16 * h + (lane & 15);
      int ka = (lane < 16) ? 0 : 8;
      U32B ab;
      ab.a = *(const uint4*)&As[buf][rr][ka];
      ab.b = *(const uint4*)&As[buf][rr][ka + 16];
      v16bf af = __builtin_bit_cast(v16bf, ab);
#pragma unroll
      for (int g2 = 0; g2 < 2; ++g2) {
        // B fragment (32x16): lanes 0-15 K{0..7,16..23}, lanes 16-31 K{8..15,24..31}
        int cc = 32 * wj + 16 * g2 + (lane & 15);
        int kb = (lane < 16) ? 0 : 8;
        U32B bb;
        bb.a = *(const uint4*)&Bs[buf][cc][kb];
        bb.b = *(const uint4*)&Bs[buf][cc][kb + 16];
        v16bf bf = __builtin_bit_cast(v16bf, bb);
        acc[h][g2] = __builtin_amdgcn_wmma_f32_16x16x32_bf16(
            false, af, false, bf, (short)0, acc[h][g2], false, false);
      }
    }
    __syncthreads();
    buf ^= 1;
  }

  // ---- epilogue: D layout = VGPR j -> (M=j | j+8), N = lane%16 ----
#pragma unroll
  for (int h = 0; h < 2; ++h)
#pragma unroll
    for (int g2 = 0; g2 < 2; ++g2) {
      int colb = col0 + 32 * wj + 16 * g2 + (lane & 15);
      int rowb = row0 + 32 * wi + 16 * h + ((lane < 16) ? 0 : 8);
#pragma unroll
      for (int j = 0; j < 8; ++j) {
        int row = rowb + j;
        float v = acc[h][g2][j];
        if constexpr (BIAS) v += bias[colb];
        if constexpr (GELU_) v = gelu_exact(v);
        size_t off = (size_t)row * M + colb;
        if constexpr (RES) v += res[off];
        if constexpr (OUT_BF16)
          ((unsigned short*)out)[off] = f2bf(v);
        else
          ((float*)out)[off] = v;
      }
    }
}

// ---------------------------------------------------------------------------
// Galerkin attention, stage 1: per-head LN of K and V (over dh=32) and the
// O(n * dh^2) reduction ctx[b,h,d,e] = sum_n k_ln[n,d] * v_ln[n,e].
// Grid (splits=16, heads=8, batch=4); each block covers 1024 tokens and
// atomically accumulates its 32x32 partial into global ctx.
// ---------------------------------------------------------------------------
__global__ __launch_bounds__(256) void k_attn_ctx(
    const unsigned short* __restrict__ qkv,  // [NT, 768]  (q|k|v)
    const float* __restrict__ kg, const float* __restrict__ kb,
    const float* __restrict__ vg, const float* __restrict__ vb,
    float* __restrict__ ctx) {               // [B, NH, 32, 32]
  __shared__ unsigned short ks[128][DH];
  __shared__ unsigned short vs[128][DH];
  const int tid   = threadIdx.x;
  const int split = blockIdx.x;
  const int head  = blockIdx.y;
  const int batch = blockIdx.z;
  const int n0    = batch * NPB + split * 1024;

  const int dd = tid >> 3;          // this thread's d (0..31)
  const int e0 = (tid & 7) * 4;     // this thread's 4 e's
  float a0 = 0.f, a1 = 0.f, a2 = 0.f, a3 = 0.f;

  for (int c = 0; c < 8; ++c) {
    {
      int tok = tid & 127;
      bool doV = tid >= 128;
      const unsigned short* src =
          qkv + (size_t)(n0 + c * 128 + tok) * 768 + 256 + (doV ? 256 : 0) + head * DH;
      float xv[DH];
      float s = 0.0f;
#pragma unroll
      for (int d = 0; d < DH; ++d) { xv[d] = bf2f(src[d]); s += xv[d]; }
      float mean = s * (1.0f / 32.0f);
      float q = 0.0f;
#pragma unroll
      for (int d = 0; d < DH; ++d) { float dv = xv[d] - mean; q += dv * dv; }
      float rs = rsqrtf(q * (1.0f / 32.0f) + 1e-5f);
      const float* gg = doV ? vg : kg;
      const float* bb = doV ? vb : kb;
      unsigned short* dst = doV ? &vs[tok][0] : &ks[tok][0];
#pragma unroll
      for (int d = 0; d < DH; ++d)
        dst[d] = f2bf((xv[d] - mean) * rs * gg[d] + bb[d]);
    }
    __syncthreads();
    for (int t = 0; t < 128; ++t) {
      float kv = bf2f(ks[t][dd]);
      a0 += kv * bf2f(vs[t][e0 + 0]);
      a1 += kv * bf2f(vs[t][e0 + 1]);
      a2 += kv * bf2f(vs[t][e0 + 2]);
      a3 += kv * bf2f(vs[t][e0 + 3]);
    }
    __syncthreads();
  }
  float* cp = ctx + (((size_t)batch * NH + head) * DH + dd) * DH + e0;
  atomicAdd(cp + 0, a0);
  atomicAdd(cp + 1, a1);
  atomicAdd(cp + 2, a2);
  atomicAdd(cp + 3, a3);
}

// ---------------------------------------------------------------------------
// Attention stage 2: o[n, h*32+e] = (1/NPB) * sum_d q[n,h,d] * ctx[b,h,d,e]
// ---------------------------------------------------------------------------
__global__ __launch_bounds__(256) void k_attn_o(
    const unsigned short* __restrict__ qkv, const float* __restrict__ ctx,
    unsigned short* __restrict__ o) {
  __shared__ float cs[NH][DH][DH];
  const int tid = threadIdx.x;
  const int n = blockIdx.x * 256 + tid;
  const int batch = n / NPB;
  const float* csrc = ctx + (size_t)batch * NH * DH * DH;
  for (int i = tid; i < NH * DH * DH; i += 256) ((float*)cs)[i] = csrc[i];
  __syncthreads();
  const float invN = 1.0f / (float)NPB;
  const unsigned short* qp = qkv + (size_t)n * 768;
  for (int h = 0; h < NH; ++h) {
    float q[DH];
#pragma unroll
    for (int d = 0; d < DH; ++d) q[d] = bf2f(qp[h * DH + d]);
    for (int e = 0; e < DH; ++e) {
      float a = 0.0f;
#pragma unroll
      for (int d = 0; d < DH; ++d) a += q[d] * cs[h][d][e];
      o[(size_t)n * DMODEL + h * DH + e] = f2bf(a * invN);
    }
  }
}

// ---------------------------------------------------------------------------
// Final projection 512 -> 3 and NHWC->NCHW write-out.
// ---------------------------------------------------------------------------
__global__ __launch_bounds__(256) void k_final(
    const unsigned short* __restrict__ hidden,  // [NT, 512] bf16
    const float* __restrict__ w2,               // [512, 3]
    const float* __restrict__ b2, float* __restrict__ out) {
  int n = blockIdx.x * 256 + threadIdx.x;
  int batch = n / NPB, hw = n % NPB;
  const unsigned short* hp = hidden + (size_t)n * 512;
  float a0 = b2[0], a1 = b2[1], a2 = b2[2];
  for (int k = 0; k < 512; ++k) {
    float hv = bf2f(hp[k]);
    a0 += hv * w2[k * 3 + 0];
    a1 += hv * w2[k * 3 + 1];
    a2 += hv * w2[k * 3 + 2];
  }
  out[((size_t)batch * 3 + 0) * NPB + hw] = a0;
  out[((size_t)batch * 3 + 1) * NPB + hw] = a1;
  out[((size_t)batch * 3 + 2) * NPB + hw] = a2;
}

// ---------------------------------------------------------------------------
// Host-side orchestration (graph-capture safe: kernels only, all on stream).
// ---------------------------------------------------------------------------
extern "C" void kernel_launch(void* const* d_in, const int* in_sizes, int n_in,
                              void* d_out, int out_size, void* d_ws, size_t ws_size,
                              hipStream_t stream) {
  (void)in_sizes; (void)n_in; (void)out_size; (void)ws_size;
  const float* x       = (const float*)d_in[0];
  const float* coords  = (const float*)d_in[1];
  const float* embed_w = (const float*)d_in[2];
  const float* embed_b = (const float*)d_in[3];
  const float* ln1_g   = (const float*)d_in[4];
  const float* ln1_b   = (const float*)d_in[5];
  const float* ln2_g   = (const float*)d_in[6];
  const float* ln2_b   = (const float*)d_in[7];
  const float* ln3_g   = (const float*)d_in[8];
  const float* ln3_b   = (const float*)d_in[9];
  const float* qkv_w   = (const float*)d_in[10];
  const float* out_w   = (const float*)d_in[11];
  const float* out_b   = (const float*)d_in[12];
  const float* kn_g    = (const float*)d_in[13];
  const float* kn_b    = (const float*)d_in[14];
  const float* vn_g    = (const float*)d_in[15];
  const float* vn_b    = (const float*)d_in[16];
  const float* mlp_w1  = (const float*)d_in[17];
  const float* mlp_b1  = (const float*)d_in[18];
  const float* mlp_w2  = (const float*)d_in[19];
  const float* mlp_b2  = (const float*)d_in[20];
  const float* proj_w1 = (const float*)d_in[21];
  const float* proj_b1 = (const float*)d_in[22];
  const float* proj_w2 = (const float*)d_in[23];
  const float* proj_b2 = (const float*)d_in[24];

  // Workspace layout (~198 MB total)
  char* ws = (char*)d_ws;
  float*          t    = (float*)ws;                                   // 64 MB
  unsigned short* bufA = (unsigned short*)(ws + ((size_t)64  << 20));  // 96 MB (qkv / hidden)
  unsigned short* bufB = (unsigned short*)(ws + ((size_t)160 << 20));  // 32 MB (ln out / o)
  unsigned short* wbf  = (unsigned short*)(ws + ((size_t)192 << 20));  // ~4.5 MB bf16 weights
  float*          ctx  = (float*)(ws + ((size_t)197 << 20));           // 128 KB

  unsigned short* wqkv = wbf;                                // [L][768][256]
  unsigned short* wout = wqkv + (size_t)4 * 256 * 768;       // [L][256][256]
  unsigned short* wm1  = wout + (size_t)4 * 256 * 256;       // [L][512][256]
  unsigned short* wm2  = wm1  + (size_t)4 * 256 * 512;       // [L][256][512]
  unsigned short* wp1  = wm2  + (size_t)4 * 512 * 256;       // [512][256]

  // Convert weights to bf16, transposed to [M][K] per layer
  auto convT = [&](const float* s, unsigned short* d, int K, int M, int L) {
    int total = K * M * L;
    k_f32_to_bf16_T<<<(total + 255) / 256, 256, 0, stream>>>(s, d, K, M, total);
  };
  convT(qkv_w,   wqkv, 256, 768, 4);
  convT(out_w,   wout, 256, 256, 4);
  convT(mlp_w1,  wm1,  256, 512, 4);
  convT(mlp_w2,  wm2,  512, 256, 4);
  convT(proj_w1, wp1,  256, 512, 1);

  // Embedding
  k_embed<<<NT, 256, 0, stream>>>(x, coords, embed_w, embed_b, t);

  for (int i = 0; i < 4; ++i) {
    // LN1 -> bf16
    k_layernorm<true><<<NT / 8, 256, 0, stream>>>(t, ln1_g + i * 256, ln1_b + i * 256, bufB);
    // QKV GEMM (no bias), bf16 out
    k_gemm_bf16<false, false, false, true><<<dim3(768 / 64, NT / 128), 256, 0, stream>>>(
        bufB, wqkv + (size_t)i * 256 * 768, nullptr, nullptr, bufA, 256, 768);
    // Galerkin attention
    k_zero_f32<<<(4 * NH * DH * DH + 255) / 256, 256, 0, stream>>>(ctx, 4 * NH * DH * DH);
    k_attn_ctx<<<dim3(16, NH, 4), 256, 0, stream>>>(
        bufA, kn_g + i * DH, kn_b + i * DH, vn_g + i * DH, vn_b + i * DH, ctx);
    k_attn_o<<<NT / 256, 256, 0, stream>>>(bufA, ctx, bufB);
    // out projection + residual (f32 into t)
    k_gemm_bf16<true, false, true, false><<<dim3(256 / 64, NT / 128), 256, 0, stream>>>(
        bufB, wout + (size_t)i * 256 * 256, out_b + i * 256, t, t, 256, 256);
    // LN2 -> bf16
    k_layernorm<true><<<NT / 8, 256, 0, stream>>>(t, ln2_g + i * 256, ln2_b + i * 256, bufB);
    // MLP1 + GELU -> bf16 hidden
    k_gemm_bf16<true, true, false, true><<<dim3(512 / 64, NT / 128), 256, 0, stream>>>(
        bufB, wm1 + (size_t)i * 256 * 512, mlp_b1 + i * 512, nullptr, bufA, 256, 512);
    // MLP2 + residual (f32 into t, pre-LN3)
    k_gemm_bf16<true, false, true, false><<<dim3(256 / 64, NT / 128), 256, 0, stream>>>(
        bufA, wm2 + (size_t)i * 512 * 256, mlp_b2 + i * 256, t, t, 512, 256);
    // LN3 in-place f32
    k_layernorm<false><<<NT / 8, 256, 0, stream>>>(t, ln3_g + i * 256, ln3_b + i * 256, t);
  }

  // Final head: proj1 (GELU) via WMMA, proj2 (512->3) via VALU
  k_f32_to_bf16<<<(NT * 256 + 255) / 256, 256, 0, stream>>>(t, bufB, NT * 256);
  k_gemm_bf16<true, true, false, true><<<dim3(512 / 64, NT / 128), 256, 0, stream>>>(
      bufB, wp1, proj_b1, nullptr, bufA, 256, 512);
  k_final<<<NT / 256, 256, 0, stream>>>(bufA, proj_w2, proj_b2, (float*)d_out);
}
